// LSTM_88459146428571
// MI455X (gfx1250) — matrix operand order repaired and checked by reading, not compiled, not run on hardware
//
#include <hip/hip_runtime.h>
#include <hip/hip_bf16.h>

typedef __attribute__((ext_vector_type(16))) _Float16 v16h;
typedef __attribute__((ext_vector_type(8)))  float    v8f;

#define S_LEN 4096
#define B_TOT 1024
#define WAVES_PER_BLK 8
#define NBLK 8   // 8 blocks * 8 waves * 16 batch rows = 1024 = B

__device__ __forceinline__ float fexp2(float x) { return __builtin_amdgcn_exp2f(x); }
__device__ __forceinline__ float frcp(float x)  { return __builtin_amdgcn_rcpf(x); }

#if __has_builtin(__builtin_amdgcn_tanhf)
__device__ __forceinline__ float ftanh(float x) { return __builtin_amdgcn_tanhf(x); }
__device__ __forceinline__ float fsigmoid(float x) {
  return __builtin_fmaf(0.5f, __builtin_amdgcn_tanhf(0.5f * x), 0.5f);
}
#else
__device__ __forceinline__ float ftanh(float x) {
  // tanh(x) = 1 - 2/(exp(2x)+1), exp via v_exp_f32 (exp2)
  float e = fexp2(x * 2.8853900817779268f);   // 2*log2(e)
  return 1.0f - 2.0f * frcp(e + 1.0f);
}
__device__ __forceinline__ float fsigmoid(float x) {
  float e = fexp2(x * -1.4426950408889634f);  // -log2(e)
  return frcp(1.0f + e);
}
#endif

__global__ __launch_bounds__(256)
void lstm_fused_wmma(const float* __restrict__ x,
                     const float* __restrict__ W_ih,
                     const float* __restrict__ W_hh,
                     const float* __restrict__ b_ih,
                     const float* __restrict__ b_hh,
                     const float* __restrict__ W_lin,
                     const float* __restrict__ b_lin,
                     float* __restrict__ out)
{
  __shared__ float hstage[WAVES_PER_BLK][16 * 16];   // per-wave private h tile [batch][hidden]
  const int lane = threadIdx.x & 31;
  const int wv   = threadIdx.x >> 5;
  const int row  = lane & 15;          // lane-within-half
  const int hi   = lane >> 4;          // 0: lanes 0-15, 1: lanes 16-31
  const int b0   = (blockIdx.x * WAVES_PER_BLK + wv) * 16;
  float* hs = hstage[wv];

  // ---- Build constant B matrices (one per gate: i,f,g,o) ----
  // B layout (16-bit 32x16): elems 0-7 hold K=8*hi+0..7 (W_hh rows, transposed),
  // elems 8-15 hold K=16+8*hi+0..7: K16,K17 = W_ih columns, K18 = b_ih+b_hh, rest 0.
  v16h bm[4];
  #pragma unroll
  for (int g = 0; g < 4; ++g) {
    v16h w;
    #pragma unroll
    for (int k = 0; k < 16; ++k) w[k] = (_Float16)0.0f;
    const int gr = g * 16 + row;       // global gate row (0..63)
    #pragma unroll
    for (int k = 0; k < 8; ++k) w[k] = (_Float16)W_hh[gr * 16 + hi * 8 + k];
    if (hi == 0) {
      w[8] = (_Float16)W_ih[gr * 2 + 0];
      w[9] = (_Float16)W_ih[gr * 2 + 1];
      w[10] = (_Float16)(b_ih[gr] + b_hh[gr]);
    }
    bm[g] = w;
  }

  // Output projection weights: each half-lane handles 8 hidden components.
  float wl[8];
  #pragma unroll
  for (int k = 0; k < 8; ++k) wl[k] = W_lin[hi * 8 + k];
  const float blin = b_lin[0];

  // h_{-1} = 0 in the LDS stage
  #pragma unroll
  for (int r = 0; r < 8; ++r) hs[(hi * 8 + r) * 16 + row] = 0.0f;

  // c state in fp32, D-layout (lane: hidden col 'row', batch rows hi*8+r)
  v8f c;
  #pragma unroll
  for (int r = 0; r < 8; ++r) c[r] = 0.0f;

  const float2* x2 = (const float2*)x;           // x[s][b][0..1]
  float2 xv = x2[0 * B_TOT + b0 + row];          // prefetch step 0
  const float hmask = (hi == 0) ? 1.0f : 0.0f;   // K=16..18 live only in lanes 0-15

  v8f cz;
  #pragma unroll
  for (int r = 0; r < 8; ++r) cz[r] = 0.0f;

  #pragma unroll 1
  for (int s = 0; s < S_LEN; ++s) {
    // prefetch x for step s+1 (hidden across the whole body)
    const int sn = (s + 1 < S_LEN) ? (s + 1) : s;
    float2 xn = x2[sn * B_TOT + b0 + row];

    // transposed read of h_{s-1}: lane gets 8 consecutive hidden components of one batch row
    const float4* hp = (const float4*)(hs + row * 16 + hi * 8);
    float4 u0 = hp[0];
    float4 u1 = hp[1];

    // fused output projection for step s-1 (reuses transposed h already in regs)
    float part = u0.x * wl[0] + u0.y * wl[1] + u0.z * wl[2] + u0.w * wl[3]
               + u1.x * wl[4] + u1.y * wl[5] + u1.z * wl[6] + u1.w * wl[7];
    part += __shfl_xor(part, 16, 32);            // combine hidden halves
    if (s > 0 && hi == 0) out[(s - 1) * B_TOT + b0 + row] = part + blin;

    // A matrix (16-bit 16x32): K0-15 = h (f16), K16/17 = x0/x1, K18 = 1.0, rest 0
    v16h a;
    a[0] = (_Float16)u0.x; a[1] = (_Float16)u0.y; a[2] = (_Float16)u0.z; a[3] = (_Float16)u0.w;
    a[4] = (_Float16)u1.x; a[5] = (_Float16)u1.y; a[6] = (_Float16)u1.z; a[7] = (_Float16)u1.w;
    a[8]  = (_Float16)(xv.x * hmask);
    a[9]  = (_Float16)(xv.y * hmask);
    a[10] = (_Float16)hmask;
    a[11] = (_Float16)0.0f; a[12] = (_Float16)0.0f; a[13] = (_Float16)0.0f;
    a[14] = (_Float16)0.0f; a[15] = (_Float16)0.0f;

    // gates[16 batch,16 gate] = [h | x | 1] x [W_hh^T ; W_ih^T ; b] , f32 accumulate
    v8f gi = __builtin_amdgcn_wmma_f32_16x16x32_f16(false, a, false, bm[0], (short)0, cz, false, false);
    v8f gf = __builtin_amdgcn_wmma_f32_16x16x32_f16(false, a, false, bm[1], (short)0, cz, false, false);
    v8f gg = __builtin_amdgcn_wmma_f32_16x16x32_f16(false, a, false, bm[2], (short)0, cz, false, false);
    v8f go = __builtin_amdgcn_wmma_f32_16x16x32_f16(false, a, false, bm[3], (short)0, cz, false, false);

    // pointwise LSTM cell in fp32 (D-layout), write new h back to LDS stage
    #pragma unroll
    for (int r = 0; r < 8; ++r) {
      float iv = fsigmoid(gi[r]);
      float fv = fsigmoid(gf[r]);
      float gv = ftanh(gg[r]);
      float ov = fsigmoid(go[r]);
      float cc = fv * c[r] + iv * gv;
      c[r] = cc;
      hs[(hi * 8 + r) * 16 + row] = ov * ftanh(cc);   // [batch][hidden]
    }
    xv = xn;
  }

  // output for the final step
  {
    const float4* hp = (const float4*)(hs + row * 16 + hi * 8);
    float4 u0 = hp[0];
    float4 u1 = hp[1];
    float part = u0.x * wl[0] + u0.y * wl[1] + u0.z * wl[2] + u0.w * wl[3]
               + u1.x * wl[4] + u1.y * wl[5] + u1.z * wl[6] + u1.w * wl[7];
    part += __shfl_xor(part, 16, 32);
    if (hi == 0) out[(S_LEN - 1) * B_TOT + b0 + row] = part + blin;
  }
}

extern "C" void kernel_launch(void* const* d_in, const int* in_sizes, int n_in,
                              void* d_out, int out_size, void* d_ws, size_t ws_size,
                              hipStream_t stream) {
  const float* x     = (const float*)d_in[0];
  const float* W_ih  = (const float*)d_in[1];
  const float* W_hh  = (const float*)d_in[2];
  const float* b_ih  = (const float*)d_in[3];
  const float* b_hh  = (const float*)d_in[4];
  const float* W_lin = (const float*)d_in[5];
  const float* b_lin = (const float*)d_in[6];
  float* out = (float*)d_out;
  (void)in_sizes; (void)n_in; (void)out_size; (void)d_ws; (void)ws_size;

  lstm_fused_wmma<<<dim3(NBLK), dim3(256), 0, stream>>>(
      x, W_ih, W_hh, b_ih, b_hh, W_lin, b_lin, out);
}
